// GptOssExperts_24507083391443
// MI455X (gfx1250) — matrix operand order repaired and checked by reading, not compiled, not run on hardware
//
#include <hip/hip_runtime.h>

// ---------------------------------------------------------------------------
// GPT-OSS MoE experts (ragged two-GEMM FFN + clipped-SiLU GLU) for gfx1250.
//   Prologue: convert x -> bf16; convert+transpose w_gate_up / w_down -> bf16
//             [n][k] layouts so GEMM staging is a pure strided 2D copy.
//   GEMM1:    hidden = act(x @ w_gate_up[e] + b_gate_up[e]) -> bf16 (d_ws)
//   GEMM2:    out    = hidden @ w_down[e] + b_down[e]       -> f32 (d_out)
// Tiles are staged into LDS by the Tensor Data Mover (TENSOR_LOAD_TO_LDS,
// TENSORcnt-pipelined, double buffered); compute is v_wmma_f32_16x16x32_bf16.
// ---------------------------------------------------------------------------

#define E_CONST 16
#define D_CONST 1024
#define M_CONST 1024
#define ALPHA_C 1.702f
#define LIMIT_C 7.0f
#define KLD 36   // LDS k-stride in bf16 elems (32 + 4 pad) -> 72B padded rows

#if defined(__has_builtin)
#if __has_builtin(__builtin_amdgcn_tensor_load_to_lds) && \
    __has_builtin(__builtin_amdgcn_s_wait_tensorcnt)
#define USE_TDM 1
#endif
#endif

typedef __attribute__((ext_vector_type(16))) __bf16 v16bf;
typedef __attribute__((ext_vector_type(8)))  float  v8f;
typedef __attribute__((ext_vector_type(4)))  unsigned int v4u;
typedef __attribute__((ext_vector_type(8)))  int    v8i;
typedef __attribute__((ext_vector_type(4)))  int    v4i;

__device__ __forceinline__ unsigned short f2bf(float f) {
  unsigned int u = __builtin_bit_cast(unsigned int, f);
  u += 0x7FFFu + ((u >> 16) & 1u);          // round-to-nearest-even
  return (unsigned short)(u >> 16);
}
__device__ __forceinline__ unsigned int pk2(float a, float b) {
  return (unsigned int)f2bf(a) | ((unsigned int)f2bf(b) << 16);
}

// A-matrix 16x32 bf16 fragment (ISA 7.12.2): lane m=lane&15, half=lane>>4;
// VGPR v holds K = (v>>2)*16 + half*8 + (v&3)*2, +1.
__device__ __forceinline__ v16bf frag_A(const unsigned short* s, int row_base, int lane) {
  const int m = lane & 15, h = lane >> 4;
  const unsigned short* rowp = s + (row_base + m) * KLD;
  v16bf f;
  unsigned int* fp = (unsigned int*)&f;
#pragma unroll
  for (int v = 0; v < 8; ++v) {
    const int k = ((v >> 2) << 4) + (h << 3) + ((v & 3) << 1);
    fp[v] = *(const unsigned int*)(rowp + k);
  }
  return f;
}

// B-matrix 32x16 bf16 fragment: same K order as A: VGPR v holds K = (v>>2)*16 + half*8 + (v&3)*2, +1;
// LDS tile is [n][k] so (k,k+1) pairs are contiguous.
__device__ __forceinline__ v16bf frag_B(const unsigned short* s, int col_base, int lane) {
  const int n = lane & 15, h = lane >> 4;
  const unsigned short* colp = s + (col_base + n) * KLD;
  v16bf f;
  unsigned int* fp = (unsigned int*)&f;
#pragma unroll
  for (int v = 0; v < 8; ++v) {
    const int k = ((v >> 2) << 4) + (h << 3) + ((v & 3) << 1);
    fp[v] = *(const unsigned int*)(colp + k);
  }
  return f;
}

__device__ __forceinline__ v8f wmma_bf(v16bf a, v16bf b, v8f c) {
  return __builtin_amdgcn_wmma_f32_16x16x32_bf16(false, a, false, b, (short)0, c, false, false);
}

#ifdef USE_TDM
// Issue one TDM 2D tile load: tile_d1 rows x tile_d0 bf16 elements, source row
// stride `stride0` elements, dest LDS rows padded 64B->72B (KLD layout).
// D# per CDNA5 ISA (§8.3/8.4). Rows/cols beyond tensor_d* read 0.
__device__ __forceinline__ void tdm_load_2d(unsigned int lds_addr,
                                            const unsigned short* gaddr,
                                            unsigned int tile_d0, unsigned int tile_d1,
                                            unsigned long long tensor_d0,
                                            unsigned long long tensor_d1,
                                            unsigned long long stride0) {
  const unsigned long long ga = (unsigned long long)(uintptr_t)gaddr;
  v4u g0;
  g0[0] = 1u;                                              // count=1 (valid user D#)
  g0[1] = lds_addr;                                        // [63:32] lds byte addr
  g0[2] = (unsigned int)ga;                                // global_addr lo
  g0[3] = (unsigned int)((ga >> 32) & 0x1FFFFFFu) | (2u << 30);  // ga hi | type=2
  const unsigned int flags = (1u << 16)   // data_size = 1 -> 2 bytes
                           | (1u << 20)   // pad_enable
                           | (3u << 22)   // pad_interval: 2^3 * 8B = 64B
                           | (1u << 25);  // pad_amount: 2 dwords = 8B
  v8i g1;
  g1[0] = (int)flags;                                          // wg_mask=0
  g1[1] = (int)(((unsigned int)tensor_d0 & 0xFFFFu) << 16);    // dim0[15:0]
  g1[2] = (int)((((unsigned int)(tensor_d0 >> 16)) & 0xFFFFu) |
                (((unsigned int)tensor_d1 & 0xFFFFu) << 16));  // dim0 hi | dim1 lo
  g1[3] = (int)((((unsigned int)(tensor_d1 >> 16)) & 0xFFFFu) |
                (tile_d0 << 16));                              // dim1 hi | tile_dim0
  g1[4] = (int)(tile_d1 & 0xFFFFu);                            // tile_dim1 (dim2=0)
  g1[5] = (int)(unsigned int)stride0;                          // stride0 lo32
  g1[6] = (int)((unsigned int)(stride0 >> 32) & 0xFFFFu);      // stride0 hi16
  g1[7] = 0;
  const v4i z4 = {0, 0, 0, 0};
#if __clang_major__ >= 23
  const v8i z8 = {0, 0, 0, 0, 0, 0, 0, 0};
  __builtin_amdgcn_tensor_load_to_lds(g0, g1, z4, z4, z8, 0);
#else
  __builtin_amdgcn_tensor_load_to_lds(g0, g1, z4, z4, 0);
#endif
}
#endif

// Map blockIdx.x -> (expert, row0, rows) over the ragged row space.
__device__ __forceinline__ void find_tile(const int* __restrict__ gs, int tile,
                                          int& expert, int& row0, int& rows) {
  int base = 0, acct = 0;
  expert = -1;
  for (int e = 0; e < E_CONST; ++e) {
    int g = gs[e];
    int nt = (g + 127) >> 7;
    if (tile < acct + nt) {
      expert = e;
      int lt = tile - acct;
      row0 = base + (lt << 7);
      int rem = base + g - row0;
      rows = rem < 128 ? rem : 128;
      return;
    }
    acct += nt;
    base += g;
  }
}

// ---------------------------------------------------------------------------
// Prologue 1: x (f32) -> bf16, straight copy.
// ---------------------------------------------------------------------------
__global__ __launch_bounds__(256)
void cvt_x_kernel(const float* __restrict__ x, unsigned short* __restrict__ xb,
                  long long n) {
  const long long step = (long long)gridDim.x * blockDim.x * 8;
  for (long long i = ((long long)blockIdx.x * blockDim.x + threadIdx.x) * 8;
       i < n; i += step) {
    const float4 a = *(const float4*)(x + i);
    const float4 b = *(const float4*)(x + i + 4);
    uint4 o;
    o.x = pk2(a.x, a.y); o.y = pk2(a.z, a.w);
    o.z = pk2(b.x, b.y); o.w = pk2(b.z, b.w);
    *(uint4*)(xb + i) = o;
  }
}

// ---------------------------------------------------------------------------
// Prologue 2: per-expert transpose+convert  src[R][C] f32 -> dst[C][R] bf16.
// grid = (C/32, R/32, E), 256 threads, 32x32 LDS tile.
// ---------------------------------------------------------------------------
__global__ __launch_bounds__(256)
void tr_cvt_kernel(const float* __restrict__ src, unsigned short* __restrict__ dst,
                   int R, int C) {
  __shared__ float t[32][33];
  const float* s = src + (size_t)blockIdx.z * R * C;
  unsigned short* d = dst + (size_t)blockIdx.z * R * C;
  const int c0 = blockIdx.x << 5, r0 = blockIdx.y << 5;
  const int tr = threadIdx.x >> 3, c4 = (threadIdx.x & 7) << 2;
  const float4 v = *(const float4*)(s + (size_t)(r0 + tr) * C + c0 + c4);
  t[tr][c4 + 0] = v.x; t[tr][c4 + 1] = v.y;
  t[tr][c4 + 2] = v.z; t[tr][c4 + 3] = v.w;
  __syncthreads();
  const unsigned int p0 = pk2(t[c4 + 0][tr], t[c4 + 1][tr]);
  const unsigned int p1 = pk2(t[c4 + 2][tr], t[c4 + 3][tr]);
  *(uint2*)(d + (size_t)(c0 + tr) * R + r0 + c4) = make_uint2(p0, p1);
}

// ---------------------------------------------------------------------------
// Stage 1: 128 rows x 64 hidden cols/block; gate AND up columns of the same
// hidden tile so the GLU fuses in registers. Double-buffered LDS.
// ---------------------------------------------------------------------------
__global__ __launch_bounds__(256)
void moe_gateup_kernel(const unsigned short* __restrict__ xb,
                       const unsigned short* __restrict__ wgut, // [E][2M][D] bf16
                       const float* __restrict__ bgu,
                       const int* __restrict__ gs,
                       unsigned short* __restrict__ hid,
                       int T) {
  __shared__ unsigned short sA[2][128 * KLD];  // x rows   [row][k]
  __shared__ unsigned short sB[2][128 * KLD];  // rows 0-63 gate, 64-127 up [n][k]

  int expert, row0, rows;
  find_tile(gs, (int)blockIdx.x, expert, row0, rows);
  if (expert < 0) return;

  const int tid  = threadIdx.x;
  const int lane = tid & 31;
  const int wave = tid >> 5;
  const int wr = (wave >> 1) << 5;          // 0,32,64,96
  const int wc = (wave & 1) << 5;           // 0,32
  const int n0 = blockIdx.y * 64;
  const int NS = D_CONST / 32;

  const v8f vzero = {0.f, 0.f, 0.f, 0.f, 0.f, 0.f, 0.f, 0.f};
  v8f gacc[2][2], uacc[2][2];
#pragma unroll
  for (int i = 0; i < 2; ++i)
#pragma unroll
    for (int j = 0; j < 2; ++j) { gacc[i][j] = vzero; uacc[i][j] = vzero; }

  const unsigned short* xg = xb + (size_t)row0 * D_CONST;
  const unsigned short* gg = wgut + ((size_t)expert * 2 * M_CONST + n0) * D_CONST;
  const unsigned short* ug = gg + (size_t)M_CONST * D_CONST;

#ifdef USE_TDM
  const unsigned int ldsA[2] = {(unsigned int)(uintptr_t)&sA[0][0],
                                (unsigned int)(uintptr_t)&sA[1][0]};
  const unsigned int ldsB[2] = {(unsigned int)(uintptr_t)&sB[0][0],
                                (unsigned int)(uintptr_t)&sB[1][0]};
  const bool issuer = (wave == 0);
#define S1_ISSUE(s)                                                            \
  do {                                                                         \
    const int kb_ = (s) << 5;                                                  \
    tdm_load_2d(ldsA[(s) & 1], xg + kb_, 32, 128,                              \
                (unsigned long long)(D_CONST - kb_),                           \
                (unsigned long long)(T - row0), D_CONST);                      \
    tdm_load_2d(ldsB[(s) & 1], gg + kb_, 32, 64,                               \
                (unsigned long long)(D_CONST - kb_), 64ull, D_CONST);          \
    tdm_load_2d(ldsB[(s) & 1] + 64 * KLD * 2, ug + kb_, 32, 64,                \
                (unsigned long long)(D_CONST - kb_), 64ull, D_CONST);          \
  } while (0)
  if (issuer) S1_ISSUE(0);
#else
#define S1_STAGE(s)                                                            \
  do {                                                                         \
    const int kb_ = (s) << 5;                                                  \
    unsigned short* dA = &sA[(s) & 1][0];                                      \
    unsigned short* dB = &sB[(s) & 1][0];                                      \
    _Pragma("unroll")                                                          \
    for (int i = 0; i < 2; ++i) {                                              \
      const int lin = tid + (i << 8);                                          \
      const int rr = lin >> 2, kq = (lin & 3) << 3;                            \
      int gr = row0 + rr; if (gr >= T) gr = T - 1;                             \
      const uint4 v = *(const uint4*)(xg + ((size_t)(gr - row0)) * D_CONST +   \
                                      kb_ + kq);                               \
      *(uint2*)&dA[rr * KLD + kq]     = make_uint2(v.x, v.y);                  \
      *(uint2*)&dA[rr * KLD + kq + 4] = make_uint2(v.z, v.w);                  \
      const unsigned short* wsrc = (rr < 64) ? (gg + (size_t)rr * D_CONST)     \
                                             : (ug + (size_t)(rr - 64) * D_CONST); \
      const uint4 w = *(const uint4*)(wsrc + kb_ + kq);                        \
      *(uint2*)&dB[rr * KLD + kq]     = make_uint2(w.x, w.y);                  \
      *(uint2*)&dB[rr * KLD + kq + 4] = make_uint2(w.z, w.w);                  \
    }                                                                          \
  } while (0)
  S1_STAGE(0);
#endif

  for (int s = 0; s < NS; ++s) {
#ifdef USE_TDM
    if (issuer) __builtin_amdgcn_s_wait_tensorcnt(0);
#endif
    __syncthreads();
#ifdef USE_TDM
    if (issuer && (s + 1 < NS)) S1_ISSUE(s + 1);
#else
    if (s + 1 < NS) S1_STAGE(s + 1);
#endif
    const unsigned short* cA = &sA[s & 1][0];
    const unsigned short* cB = &sB[s & 1][0];
    const v16bf a0  = frag_A(cA, wr, lane);
    const v16bf a1  = frag_A(cA, wr + 16, lane);
    const v16bf bg0 = frag_B(cB, wc, lane);
    const v16bf bg1 = frag_B(cB, wc + 16, lane);
    const v16bf bu0 = frag_B(cB, 64 + wc, lane);
    const v16bf bu1 = frag_B(cB, 64 + wc + 16, lane);
    gacc[0][0] = wmma_bf(a0, bg0, gacc[0][0]);
    gacc[0][1] = wmma_bf(a0, bg1, gacc[0][1]);
    gacc[1][0] = wmma_bf(a1, bg0, gacc[1][0]);
    gacc[1][1] = wmma_bf(a1, bg1, gacc[1][1]);
    uacc[0][0] = wmma_bf(a0, bu0, uacc[0][0]);
    uacc[0][1] = wmma_bf(a0, bu1, uacc[0][1]);
    uacc[1][0] = wmma_bf(a1, bu0, uacc[1][0]);
    uacc[1][1] = wmma_bf(a1, bu1, uacc[1][1]);
  }

  // epilogue: bias + clip + GLU -> bf16 hidden
  const float* bge = bgu + (size_t)expert * (2 * M_CONST);
  const int h = lane >> 4, nl = lane & 15;
#pragma unroll
  for (int j = 0; j < 2; ++j) {
    const int col = n0 + wc + (j << 4) + nl;
    const float bg = bge[col];
    const float bu = bge[M_CONST + col];
#pragma unroll
    for (int i = 0; i < 2; ++i) {
#pragma unroll
      for (int r = 0; r < 8; ++r) {
        const int rl = wr + (i << 4) + (h << 3) + r;   // C/D: row = r + 8*half
        if (rl < rows) {
          float gv = gacc[i][j][r] + bg;
          gv = fminf(fmaxf(gv, -LIMIT_C), LIMIT_C);
          float uv = uacc[i][j][r] + bu;
          uv = fminf(fmaxf(uv, -LIMIT_C), LIMIT_C);
          const float sg = __builtin_amdgcn_rcpf(1.0f + __expf(-ALPHA_C * gv));
          const float hv = (uv + 1.0f) * (gv * sg);
          hid[(size_t)(row0 + rl) * M_CONST + col] = f2bf(hv);
        }
      }
    }
  }
}

// ---------------------------------------------------------------------------
// Stage 2: out = hidden @ w_down[e] + b_down[e]; 128x128 block tiles.
// ---------------------------------------------------------------------------
__global__ __launch_bounds__(256)
void moe_down_kernel(const unsigned short* __restrict__ hid,
                     const unsigned short* __restrict__ wdt,  // [E][D][M] bf16
                     const float* __restrict__ bdn,
                     const int* __restrict__ gs,
                     float* __restrict__ out,
                     int T) {
  __shared__ unsigned short sA[2][128 * KLD];
  __shared__ unsigned short sB[2][128 * KLD];

  int expert, row0, rows;
  find_tile(gs, (int)blockIdx.x, expert, row0, rows);
  if (expert < 0) return;

  const int tid  = threadIdx.x;
  const int lane = tid & 31;
  const int wave = tid >> 5;
  const int wr = (wave >> 1) << 5;          // 0,32,64,96
  const int wc = (wave & 1) << 6;           // 0,64
  const int n0 = blockIdx.y << 7;
  const int NS = M_CONST / 32;

  const v8f vzero = {0.f, 0.f, 0.f, 0.f, 0.f, 0.f, 0.f, 0.f};
  v8f acc[2][4];
#pragma unroll
  for (int i = 0; i < 2; ++i)
#pragma unroll
    for (int j = 0; j < 4; ++j) acc[i][j] = vzero;

  const unsigned short* ag = hid + (size_t)row0 * M_CONST;
  const unsigned short* bg = wdt + ((size_t)expert * D_CONST + n0) * M_CONST;

#ifdef USE_TDM
  const unsigned int ldsA[2] = {(unsigned int)(uintptr_t)&sA[0][0],
                                (unsigned int)(uintptr_t)&sA[1][0]};
  const unsigned int ldsB[2] = {(unsigned int)(uintptr_t)&sB[0][0],
                                (unsigned int)(uintptr_t)&sB[1][0]};
  const bool issuer = (wave == 0);
#define S2_ISSUE(s)                                                            \
  do {                                                                         \
    const int kb_ = (s) << 5;                                                  \
    tdm_load_2d(ldsA[(s) & 1], ag + kb_, 32, 128,                              \
                (unsigned long long)(M_CONST - kb_),                           \
                (unsigned long long)(T - row0), M_CONST);                      \
    tdm_load_2d(ldsB[(s) & 1], bg + kb_, 32, 128,                              \
                (unsigned long long)(M_CONST - kb_), 128ull, M_CONST);         \
  } while (0)
  if (issuer) S2_ISSUE(0);
#else
#define S2_STAGE(s)                                                            \
  do {                                                                         \
    const int kb_ = (s) << 5;                                                  \
    unsigned short* dA = &sA[(s) & 1][0];                                      \
    unsigned short* dB = &sB[(s) & 1][0];                                      \
    _Pragma("unroll")                                                          \
    for (int i = 0; i < 2; ++i) {                                              \
      const int lin = tid + (i << 8);                                          \
      const int rr = lin >> 2, kq = (lin & 3) << 3;                            \
      int gr = row0 + rr; if (gr >= T) gr = T - 1;                             \
      const uint4 v = *(const uint4*)(ag + ((size_t)(gr - row0)) * M_CONST +   \
                                      kb_ + kq);                               \
      *(uint2*)&dA[rr * KLD + kq]     = make_uint2(v.x, v.y);                  \
      *(uint2*)&dA[rr * KLD + kq + 4] = make_uint2(v.z, v.w);                  \
      const uint4 w = *(const uint4*)(bg + (size_t)rr * M_CONST + kb_ + kq);   \
      *(uint2*)&dB[rr * KLD + kq]     = make_uint2(w.x, w.y);                  \
      *(uint2*)&dB[rr * KLD + kq + 4] = make_uint2(w.z, w.w);                  \
    }                                                                          \
  } while (0)
  S2_STAGE(0);
#endif

  for (int s = 0; s < NS; ++s) {
#ifdef USE_TDM
    if (issuer) __builtin_amdgcn_s_wait_tensorcnt(0);
#endif
    __syncthreads();
#ifdef USE_TDM
    if (issuer && (s + 1 < NS)) S2_ISSUE(s + 1);
#else
    if (s + 1 < NS) S2_STAGE(s + 1);
#endif
    const unsigned short* cA = &sA[s & 1][0];
    const unsigned short* cB = &sB[s & 1][0];
    const v16bf a0 = frag_A(cA, wr, lane);
    const v16bf a1 = frag_A(cA, wr + 16, lane);
#pragma unroll
    for (int j = 0; j < 4; ++j) {
      const v16bf b = frag_B(cB, wc + (j << 4), lane);
      acc[0][j] = wmma_bf(a0, b, acc[0][j]);
      acc[1][j] = wmma_bf(a1, b, acc[1][j]);
    }
  }

  const float* bde = bdn + (size_t)expert * D_CONST;
  const int h = lane >> 4, nl = lane & 15;
#pragma unroll
  for (int j = 0; j < 4; ++j) {
    const int col = n0 + wc + (j << 4) + nl;
    const float bv = bde[col];
#pragma unroll
    for (int i = 0; i < 2; ++i) {
#pragma unroll
      for (int r = 0; r < 8; ++r) {
        const int rl = wr + (i << 4) + (h << 3) + r;
        if (rl < rows) {
          out[(size_t)(row0 + rl) * D_CONST + col] = acc[i][j][r] + bv;
        }
      }
    }
  }
}

// ---------------------------------------------------------------------------
extern "C" void kernel_launch(void* const* d_in, const int* in_sizes, int n_in,
                              void* d_out, int out_size, void* d_ws, size_t ws_size,
                              hipStream_t stream) {
  const float* x   = (const float*)d_in[0];
  const float* wgu = (const float*)d_in[1];
  const float* bgu = (const float*)d_in[2];
  const float* wd  = (const float*)d_in[3];
  const float* bdn = (const float*)d_in[4];
  const int*   gs  = (const int*)d_in[5];
  float* out = (float*)d_out;

  const int T = in_sizes[0] / D_CONST;

  // workspace layout (bf16): x | wgu_t [E][2M][D] | wd_t [E][D][M] | hidden [T][M]
  char* w = (char*)d_ws;
  unsigned short* xb   = (unsigned short*)w;
  unsigned short* wgut = (unsigned short*)(w + (size_t)T * D_CONST * 2);
  unsigned short* wdt  = (unsigned short*)((char*)wgut +
                          (size_t)E_CONST * 2 * M_CONST * D_CONST * 2);
  unsigned short* hid  = (unsigned short*)((char*)wdt +
                          (size_t)E_CONST * D_CONST * M_CONST * 2);

  const int row_tiles = (T + 127) / 128 + E_CONST;  // covers ragged splits
  dim3 blk(256);

  cvt_x_kernel<<<2048, blk, 0, stream>>>(x, xb, (long long)T * D_CONST);
  tr_cvt_kernel<<<dim3(2 * M_CONST / 32, D_CONST / 32, E_CONST), blk, 0, stream>>>(
      wgu, wgut, D_CONST, 2 * M_CONST);
  tr_cvt_kernel<<<dim3(M_CONST / 32, M_CONST / 32, E_CONST), blk, 0, stream>>>(
      wd, wdt, M_CONST, D_CONST);

  moe_gateup_kernel<<<dim3(row_tiles, M_CONST / 64), blk, 0, stream>>>(
      xb, wgut, bgu, gs, hid, T);
  moe_down_kernel<<<dim3(row_tiles, D_CONST / 128), blk, 0, stream>>>(
      hid, wdt, bdn, gs, out, T);
}
